// HilbertAttentionTritonWrapper_68152541053162
// MI455X (gfx1250) — compile-verified, matrix operand order checked by reading, not run on hardware
//
#include <hip/hip_runtime.h>

// ---------------------------------------------------------------------------
// Problem constants (from the reference)
//   B=2, M=16384, H=8, D=64, HID=512, SEG=128, DIL=2
//   rows = B*M = 32768
// ---------------------------------------------------------------------------

typedef __attribute__((ext_vector_type(16))) __bf16 v16bf;
typedef __attribute__((ext_vector_type(8)))  float  v8f;

union FragBF { uint4 u[2]; v16bf v; };

__device__ inline __bf16 tobf(float f) {
  union { float f; unsigned u; } a; a.f = f;
  unsigned r = a.u + 0x7FFFu + ((a.u >> 16) & 1u);   // round-to-nearest-even
  union { unsigned short s; __bf16 b; } o; o.s = (unsigned short)(r >> 16);
  return o.b;
}

// ---------------------------------------------------------------------------
// Hilbert curve tables: PERM[d] = token index of d-th curve point (128x128),
// INV[token] = curve position. Computed on device (not harness inputs).
// ---------------------------------------------------------------------------
__global__ void perm_init(int* __restrict__ perm, int* __restrict__ invp) {
  int d = blockIdx.x * 256 + threadIdx.x;
  if (d >= 16384) return;
  int t = d, x = 0, y = 0;
  for (int s = 1; s < 128; s <<= 1) {
    int rx = (t >> 1) & 1;
    int ry = (t ^ rx) & 1;
    if (ry == 0) {
      if (rx == 1) { x = s - 1 - x; y = s - 1 - y; }
      int tmp = x; x = y; y = tmp;
    }
    x += s * rx; y += s * ry;
    t >>= 2;
  }
  int pd = y * 128 + x;
  perm[d] = pd;
  invp[pd] = d;
}

// ---------------------------------------------------------------------------
// 128x64-tile WMMA GEMM, 256 threads = 8 wave32.
// Each wave owns one 16-row strip (mi = wave) and 4 column sub-tiles
// (4 accumulators) -> 4 WMMAs per wave per 32-wide K chunk, which halves the
// B-tile LDS traffic and barrier count per FLOP vs a 64x64 tile.
// bf16 operands staged in LDS (fp32 converted in flight), fp32 accumulation.
// Next K-chunk is prefetched (global_prefetch_b8) while this chunk computes.
//
// MODE 0: C = (q@Wq + k@Wk + v@Wv)/3  (A = 3 fp32 sources, K=1536), bf16 out
// MODE 1: C = X @ Wqkv, rows scattered through INV[] (Hilbert), bf16 out
// MODE 2: C = ATT @ Wout, fp32 out
// ---------------------------------------------------------------------------
template<int MODE>
__global__ __launch_bounds__(256)
void gemm_wmma(const float* __restrict__ Aq, const float* __restrict__ Ak,
               const float* __restrict__ Av, const void* __restrict__ Abfv,
               const float* __restrict__ B0, const float* __restrict__ B1,
               const float* __restrict__ B2, void* __restrict__ Cout,
               const int* __restrict__ invp, int Ntot)
{
  __shared__ alignas(16) __bf16 As[128][40];  // [m][k], rows 80B (16B-aligned)
  __shared__ alignas(16) __bf16 Bs[64][40];   // [n][k] (transposed weight tile)

  const __bf16* Abf = (const __bf16*)Abfv;
  const int tid   = threadIdx.x;
  const int mBase = blockIdx.x * 128;
  const int nBase = blockIdx.y * 64;
  const int wave  = tid >> 5;
  const int lane  = tid & 31;
  const int mi    = wave;               // 0..7 : 16-row strip
  const int ln    = lane & 15;
  const int hi    = lane >> 4;

  v8f acc[4] = {{}, {}, {}, {}};
  const int KT = (MODE == 0) ? 1536 : 512;

  // A-stage mapping: 128x32 tile, 16 elems/thread
  const int arow = tid >> 1;
  const int acb  = (tid & 1) * 16;
  // B-stage mapping: 32x64 tile, 8 elems/thread
  const int bkr  = tid >> 3;            // 0..31
  const int bnb  = (tid & 7) * 8;       // 0..56

  for (int kc = 0; kc < KT; kc += 32) {
    // ---- stage A tile 128x32 ----
    if (MODE == 0) {
      const int sseg = kc >> 9;                               // which source
      const float* Ap = (sseg == 0) ? Aq : (sseg == 1) ? Ak : Av;
      const int kk = kc & 511;
      const float4* src = (const float4*)(Ap + (size_t)(mBase + arow) * 512 + kk + acb);
      float4 x0 = src[0], x1 = src[1], x2 = src[2], x3 = src[3];
      __bf16* dst = &As[arow][acb];
      dst[0]=tobf(x0.x);  dst[1]=tobf(x0.y);  dst[2]=tobf(x0.z);  dst[3]=tobf(x0.w);
      dst[4]=tobf(x1.x);  dst[5]=tobf(x1.y);  dst[6]=tobf(x1.z);  dst[7]=tobf(x1.w);
      dst[8]=tobf(x2.x);  dst[9]=tobf(x2.y);  dst[10]=tobf(x2.z); dst[11]=tobf(x2.w);
      dst[12]=tobf(x3.x); dst[13]=tobf(x3.y); dst[14]=tobf(x3.z); dst[15]=tobf(x3.w);
    } else {
      const uint4* src = (const uint4*)(Abf + (size_t)(mBase + arow) * 512 + kc + acb);
      *(uint4*)&As[arow][acb]     = src[0];
      *(uint4*)&As[arow][acb + 8] = src[1];
    }
    // ---- stage B tile transposed: Bs[n][k] from W[k][n] ----
    {
      const float* Bp; int kk;
      if (MODE == 0) { const int sseg = kc >> 9;
                       Bp = (sseg == 0) ? B0 : (sseg == 1) ? B1 : B2;
                       kk = (kc & 511) + bkr; }
      else           { Bp = B0; kk = kc + bkr; }
      const float4* src = (const float4*)(Bp + (size_t)kk * Ntot + nBase + bnb);
      float4 x0 = src[0], x1 = src[1];
      Bs[bnb+0][bkr]=tobf(x0.x); Bs[bnb+1][bkr]=tobf(x0.y);
      Bs[bnb+2][bkr]=tobf(x0.z); Bs[bnb+3][bkr]=tobf(x0.w);
      Bs[bnb+4][bkr]=tobf(x1.x); Bs[bnb+5][bkr]=tobf(x1.y);
      Bs[bnb+6][bkr]=tobf(x1.z); Bs[bnb+7][bkr]=tobf(x1.w);
    }
    // ---- prefetch next K chunk while this one computes ----
    if (kc + 32 < KT) {
      const int kn = kc + 32;
      if (MODE == 0) {
        const int sseg = kn >> 9;
        const float* ApN = (sseg == 0) ? Aq : (sseg == 1) ? Ak : Av;
        const float* BpN = (sseg == 0) ? B0 : (sseg == 1) ? B1 : B2;
        __builtin_prefetch(ApN + (size_t)(mBase + arow) * 512 + (kn & 511) + acb, 0, 1);
        __builtin_prefetch(BpN + (size_t)((kn & 511) + bkr) * Ntot + nBase + bnb, 0, 1);
      } else {
        __builtin_prefetch(Abf + (size_t)(mBase + arow) * 512 + kn + acb, 0, 1);
        __builtin_prefetch(B0  + (size_t)(kn + bkr) * Ntot + nBase + bnb, 0, 1);
      }
    }
    __syncthreads();

    // ---- fragments per ISA 16-bit layout: K in [hi*8, +8) and [16+hi*8, +8)
    FragBF a;
    a.u[0] = *(const uint4*)&As[mi*16 + ln][hi*8];
    a.u[1] = *(const uint4*)&As[mi*16 + ln][16 + hi*8];
    #pragma unroll
    for (int ni = 0; ni < 4; ++ni) {
      FragBF b;
      b.u[0] = *(const uint4*)&Bs[ni*16 + ln][hi*8];
      b.u[1] = *(const uint4*)&Bs[ni*16 + ln][16 + hi*8];
      acc[ni] = __builtin_amdgcn_wmma_f32_16x16x32_bf16(false, a.v, false, b.v,
                                                        (short)0, acc[ni], false, false);
    }
    __syncthreads();
  }

  // ---- store C (D layout: lane = col, VGPR g -> row g + 8*hi) ----
  #pragma unroll
  for (int g = 0; g < 8; ++g) {
    const int row = mBase + mi*16 + g + 8*hi;
    if (MODE == 0) {
      __bf16* X = (__bf16*)Cout;
      #pragma unroll
      for (int ni = 0; ni < 4; ++ni)
        X[(size_t)row*512 + nBase + ni*16 + ln] = tobf(acc[ni][g] * (1.0f/3.0f));
    } else if (MODE == 1) {
      __bf16* Q = (__bf16*)Cout;
      const int bb = row >> 14, mm = row & 16383;
      const int rp = (bb << 14) + invp[mm];        // token -> curve position
      #pragma unroll
      for (int ni = 0; ni < 4; ++ni)
        Q[(size_t)rp*1536 + nBase + ni*16 + ln] = tobf(acc[ni][g]);
    } else {
      float* O = (float*)Cout;
      #pragma unroll
      for (int ni = 0; ni < 4; ++ni)
        O[(size_t)row*512 + nBase + ni*16 + ln] = acc[ni][g];
    }
  }
}

// ---------------------------------------------------------------------------
// Segmented dilated attention over Hilbert-ordered tokens.
// One block per (b, seg, dilation-phase r, head h): 64 queries x 64 keys x 64 dim.
// scores = q@k^T via WMMA (k tile already [s][d] == B's [n][k] layout),
// softmax fp32 in LDS, O = P@V via WMMA (V staged transposed [d][s]).
// Inverse Hilbert permutation folded into the output store.
// ---------------------------------------------------------------------------
__global__ __launch_bounds__(256)
void attn_wmma(const void* __restrict__ qkvv, void* __restrict__ attv,
               const int* __restrict__ perm)
{
  __shared__ alignas(16) __bf16 qs[64][72];   // q tile; reused for P (bf16)
  __shared__ alignas(16) __bf16 ks[64][72];   // k tile [s][d]
  __shared__ alignas(16) __bf16 vT[64][72];   // v transposed [d][s]
  __shared__ float sc[64][65];                // fp32 scores

  const __bf16* qkv = (const __bf16*)qkvv;
  __bf16* att = (__bf16*)attv;

  const int tid = threadIdx.x;
  const int bi  = blockIdx.x;
  const int h   = bi & 7;
  const int r   = (bi >> 3) & 1;
  const int seg = (bi >> 4) & 127;
  const int b   = bi >> 11;

  // ---- load q,k,v tiles: thread -> (t = tid/4, 16 dims) ----
  {
    const int t  = tid >> 2;
    const int db = (tid & 3) * 16;
    const int p  = seg*128 + 2*t + r;                 // curve position
    const size_t base = ((size_t)b * 16384 + p) * 1536 + h*64 + db;
    const uint4* qsrc = (const uint4*)(qkv + base);
    const uint4* ksrc = (const uint4*)(qkv + base + 512);
    const uint4* vsrc = (const uint4*)(qkv + base + 1024);
    *(uint4*)&qs[t][db]     = qsrc[0];
    *(uint4*)&qs[t][db + 8] = qsrc[1];
    *(uint4*)&ks[t][db]     = ksrc[0];
    *(uint4*)&ks[t][db + 8] = ksrc[1];
    uint4 v0 = vsrc[0], v1 = vsrc[1];
    const __bf16* va = (const __bf16*)&v0;
    const __bf16* vb = (const __bf16*)&v1;
    #pragma unroll
    for (int j = 0; j < 8; ++j) vT[db + j][t]     = va[j];
    #pragma unroll
    for (int j = 0; j < 8; ++j) vT[db + 8 + j][t] = vb[j];
  }
  __syncthreads();

  const int wave = tid >> 5, lane = tid & 31;
  const int mi = wave >> 1, ni0 = (wave & 1) * 2;
  const int ln = lane & 15, hi = lane >> 4;

  // ---- scores = (q @ k^T) * 1/sqrt(D) ----
  {
    v8f s0 = {}, s1 = {};
    #pragma unroll
    for (int kk = 0; kk < 64; kk += 32) {
      FragBF a, b0, b1;
      a.u[0]  = *(const uint4*)&qs[mi*16 + ln][kk + hi*8];
      a.u[1]  = *(const uint4*)&qs[mi*16 + ln][kk + 16 + hi*8];
      b0.u[0] = *(const uint4*)&ks[ni0*16 + ln][kk + hi*8];
      b0.u[1] = *(const uint4*)&ks[ni0*16 + ln][kk + 16 + hi*8];
      b1.u[0] = *(const uint4*)&ks[(ni0+1)*16 + ln][kk + hi*8];
      b1.u[1] = *(const uint4*)&ks[(ni0+1)*16 + ln][kk + 16 + hi*8];
      s0 = __builtin_amdgcn_wmma_f32_16x16x32_bf16(false, a.v, false, b0.v, (short)0, s0, false, false);
      s1 = __builtin_amdgcn_wmma_f32_16x16x32_bf16(false, a.v, false, b1.v, (short)0, s1, false, false);
    }
    #pragma unroll
    for (int g = 0; g < 8; ++g) {
      sc[mi*16 + g + 8*hi][ni0*16 + ln]     = s0[g] * 0.125f;
      sc[mi*16 + g + 8*hi][(ni0+1)*16 + ln] = s1[g] * 0.125f;
    }
  }
  __syncthreads();

  // ---- row-wise softmax (64 rows, one thread each), P -> qs as bf16 ----
  if (tid < 64) {
    float mx = -3.4e38f;
    for (int s = 0; s < 64; ++s) mx = fmaxf(mx, sc[tid][s]);
    float sum = 0.f;
    for (int s = 0; s < 64; ++s) { float e = __expf(sc[tid][s] - mx); sc[tid][s] = e; sum += e; }
    const float rs = 1.f / sum;
    for (int s = 0; s < 64; ++s) qs[tid][s] = tobf(sc[tid][s] * rs);
  }
  __syncthreads();

  // ---- O = P @ V ----
  v8f o0 = {}, o1 = {};
  #pragma unroll
  for (int kk = 0; kk < 64; kk += 32) {
    FragBF a, b0, b1;
    a.u[0]  = *(const uint4*)&qs[mi*16 + ln][kk + hi*8];
    a.u[1]  = *(const uint4*)&qs[mi*16 + ln][kk + 16 + hi*8];
    b0.u[0] = *(const uint4*)&vT[ni0*16 + ln][kk + hi*8];
    b0.u[1] = *(const uint4*)&vT[ni0*16 + ln][kk + 16 + hi*8];
    b1.u[0] = *(const uint4*)&vT[(ni0+1)*16 + ln][kk + hi*8];
    b1.u[1] = *(const uint4*)&vT[(ni0+1)*16 + ln][kk + 16 + hi*8];
    o0 = __builtin_amdgcn_wmma_f32_16x16x32_bf16(false, a.v, false, b0.v, (short)0, o0, false, false);
    o1 = __builtin_amdgcn_wmma_f32_16x16x32_bf16(false, a.v, false, b1.v, (short)0, o1, false, false);
  }
  #pragma unroll
  for (int g = 0; g < 8; ++g) {
    const int t  = mi*16 + g + 8*hi;
    const int p  = seg*128 + 2*t + r;
    const int m0 = perm[p];                             // curve pos -> token
    const size_t orow = ((size_t)b * 16384 + m0) * 512 + h*64;
    att[orow + ni0*16 + ln]       = tobf(o0[g]);
    att[orow + (ni0+1)*16 + ln]   = tobf(o1[g]);
  }
}

// ---------------------------------------------------------------------------
// Host launcher (graph-capture safe: stream launches only).
// Workspace: PERM/INV (128KB) + X bf16 (33.5MB) + QKV bf16 (100.7MB)
//            + ATT bf16 (33.5MB)  ~= 168 MB
// ---------------------------------------------------------------------------
extern "C" void kernel_launch(void* const* d_in, const int* in_sizes, int n_in,
                              void* d_out, int out_size, void* d_ws, size_t ws_size,
                              hipStream_t stream)
{
  (void)in_sizes; (void)n_in; (void)out_size; (void)ws_size;
  const float* q    = (const float*)d_in[0];
  const float* k    = (const float*)d_in[1];
  const float* v    = (const float*)d_in[2];
  const float* Wq   = (const float*)d_in[3];
  const float* Wk   = (const float*)d_in[4];
  const float* Wv   = (const float*)d_in[5];
  const float* Wqkv = (const float*)d_in[6];
  const float* Wout = (const float*)d_in[7];

  char* w = (char*)d_ws;
  size_t off = 0;
  auto aln = [](size_t x) { return (x + 255) & ~(size_t)255; };
  int*  perm = (int*)(w + off);  off = aln(off + (size_t)16384 * sizeof(int));
  int*  invp = (int*)(w + off);  off = aln(off + (size_t)16384 * sizeof(int));
  void* X    = (void*)(w + off); off = aln(off + (size_t)32768 * 512  * 2);
  void* QKV  = (void*)(w + off); off = aln(off + (size_t)32768 * 1536 * 2);
  void* ATT  = (void*)(w + off); off = aln(off + (size_t)32768 * 512  * 2);

  perm_init<<<dim3(64), dim3(256), 0, stream>>>(perm, invp);

  // X = (q@Wq + k@Wk + v@Wv)/3   [32768,512], bf16
  gemm_wmma<0><<<dim3(256, 8), dim3(256), 0, stream>>>(
      q, k, v, nullptr, Wq, Wk, Wv, X, nullptr, 512);

  // QKV = X @ Wqkv, Hilbert-permuted rows   [32768,1536], bf16
  gemm_wmma<1><<<dim3(256, 24), dim3(256), 0, stream>>>(
      nullptr, nullptr, nullptr, X, Wqkv, nullptr, nullptr, QKV, invp, 1536);

  // Segmented dilated attention: 2*128*2*8 = 4096 blocks
  attn_wmma<<<dim3(4096), dim3(256), 0, stream>>>(QKV, ATT, perm);

  // out = ATT @ Wout   [32768,512], fp32
  gemm_wmma<2><<<dim3(256, 8), dim3(256), 0, stream>>>(
      nullptr, nullptr, nullptr, ATT, Wout, nullptr, nullptr, d_out, nullptr, 512);
}
